// GraphAttentionLayer_83365315215886
// MI455X (gfx1250) — compile-verified, matrix operand order checked by reading, not run on hardware
//
#include <hip/hip_runtime.h>

typedef __attribute__((ext_vector_type(16))) _Float16 v16h;
typedef __attribute__((ext_vector_type(8)))  _Float16 h8;
typedef __attribute__((ext_vector_type(8)))  float    v8f;
typedef __attribute__((ext_vector_type(4)))  float    f4;
typedef __attribute__((ext_vector_type(4)))  int      i4;
typedef __attribute__((ext_vector_type(8)))  int      i8v;
typedef __attribute__((ext_vector_type(4)))  unsigned u4;

#define NEG_BIG  (-9000000000000000.0f)
#define LRELU_A  0.2f
#define LOG2E    1.44269504088896340736f
#define BATCH    8
#define NNODE    2048
#define FEAT     64
#define KPAD     40   // padded LDS row length in halfs (80 B: 16B-aligned, low-conflict)

#if defined(__gfx1250__) && __has_builtin(__builtin_amdgcn_tensor_load_to_lds)
#define HAVE_TDM 1
#else
#define HAVE_TDM 0
#endif

// ---------- fast-math helpers (guarded) ----------
__device__ __forceinline__ float fast_exp2(float x) {
#if defined(__gfx1250__) && __has_builtin(__builtin_amdgcn_exp2f)
  return __builtin_amdgcn_exp2f(x);
#else
  return exp2f(x);
#endif
}
__device__ __forceinline__ float fast_rcp(float x) {
#if defined(__gfx1250__) && __has_builtin(__builtin_amdgcn_rcpf)
  return __builtin_amdgcn_rcpf(x);
#else
  return 1.0f / x;
#endif
}

// ---------- cross-lane helpers (wave32) ----------
__device__ __forceinline__ float xor16(float v) {
  // ds_swizzle SWAPX16: offset = xor_mask(0x10)<<10 | and_mask(0x1f) = 0x401F
  int r = __builtin_amdgcn_ds_swizzle(__builtin_bit_cast(int, v), 0x401F);
  return __builtin_bit_cast(float, r);
}
__device__ __forceinline__ float lane_bcast(float v, int srcLane) {
  int r = __builtin_amdgcn_ds_bpermute(srcLane << 2, __builtin_bit_cast(int, v));
  return __builtin_bit_cast(float, r);
}

#if HAVE_TDM
// TDM: load a 64(feature) x 32(K) f16 tile of WhT into LDS with 16B row padding.
// group0: count=1 | lds_addr[63:32] | global_addr[120:64] | type=2 [127:126]
// group1: data_size=2B, pad_enable, pad_interval=64B(code 3), pad_amount=4dw(code 3),
//         tensor_dim0=2048, tensor_dim1=64, tile_dim0=32, tile_dim1=64, stride0=2048
__device__ __forceinline__ void tdm_load_wh(const _Float16* gsrc, unsigned lds_byte) {
  unsigned long long ga = (unsigned long long)(size_t)gsrc;
  u4 g0 = {};
  g0[0] = 1u;                                                 // count=1, no gather
  g0[1] = lds_byte;                                           // lds_addr
  g0[2] = (unsigned)(ga & 0xFFFFFFFFull);                     // global_addr[31:0]
  g0[3] = (unsigned)((ga >> 32) & 0x1FFFFFFull) | (2u << 30); // addr[56:32] | type=2
  i8v g1 = {};
  g1[0] = (int)((1u << 16)      // data_size = 2 bytes
              | (1u << 20)      // pad_enable
              | (3u << 22)      // pad_interval: 16 dwords = 64 B
              | (3u << 25));    // pad_amount: 4 dwords = 16 B  -> 80 B row stride
  g1[1] = (int)(2048u << 16);   // tensor_dim0[15:0] -> bits[63:48]
  g1[2] = (int)(64u << 16);     // tensor_dim1[15:0] -> bits[95:80]
  g1[3] = (int)(32u << 16);     // tile_dim0=32 -> bits[127:112]
  g1[4] = 64;                   // tile_dim1=64 (tile_dim2=0)
  g1[5] = 2048;                 // tensor_dim0_stride[31:0]
  g1[6] = 0;
  g1[7] = 0;
  i4  gz  = {};
  i8v gz8 = {};
  __builtin_amdgcn_tensor_load_to_lds(g0, g1, gz, gz, gz8, 0);
}
#endif

// ---------------- prep: Wh (f16, feature-major), s=Wh.a1, t=Wh.a2 (xLOG2E) --
__global__ void __launch_bounds__(64) gat_prep(const float* __restrict__ h,
                                               const float* __restrict__ W,
                                               const float* __restrict__ a,
                                               _Float16* __restrict__ whT,
                                               float* __restrict__ s,
                                               float* __restrict__ t) {
  const int r = blockIdx.x;          // global row 0..16383
  const int b = r >> 11;
  const int n = r & (NNODE - 1);
  const int o = threadIdx.x;         // output feature
  __shared__ float hr[FEAT];
  __shared__ float red1[FEAT];
  __shared__ float red2[FEAT];
  hr[o] = h[(size_t)r * FEAT + o];
  __syncthreads();
  float acc = 0.f;
#pragma unroll
  for (int f = 0; f < FEAT; ++f) acc = fmaf(hr[f], W[f * FEAT + o], acc);
  whT[((size_t)b * FEAT + o) * NNODE + n] = (_Float16)acc;
  red1[o] = acc * a[o];
  red2[o] = acc * a[FEAT + o];
  __syncthreads();
  if (o == 0) {
    float ss = 0.f, tt = 0.f;
#pragma unroll
    for (int i = 0; i < FEAT; ++i) { ss += red1[i]; tt += red2[i]; }
    s[r] = ss * LOG2E;   // exp2-domain: exp(e) == exp2(e*log2e)
    t[r] = tt * LOG2E;
  }
}

// ---------------- fused masked-softmax @ Wh with WMMA -----------------------
__global__ void __launch_bounds__(128) gat_main(const int* __restrict__ adj,
                                                const _Float16* __restrict__ whT,
                                                const float* __restrict__ s,
                                                const float* __restrict__ t,
                                                float* __restrict__ out) {
  __shared__ _Float16 ldsWh[2][FEAT][KPAD];  // padded B tiles : 10 KB
  __shared__ float    ldsT[NNODE];           // t[b][:]*log2e  :  8 KB

  const int tid  = threadIdx.x;
  const int wave = __builtin_amdgcn_readfirstlane(tid >> 5);  // wave-uniform scalar
  const int lane = tid & 31;
  const int m    = lane & 15;     // A row / C column within tile
  const int hi   = lane >> 4;     // K-half selector for A/B fragments
  const int off  = hi << 3;       // K base offset (0 or 8)

  const int b    = blockIdx.x >> 5;       // 32 row-blocks per batch
  const int iblk = blockIdx.x & 31;
  const int i0   = iblk * 64 + wave * 16; // this wave's first row

  for (int j = tid; j < NNODE; j += 128) ldsT[j] = t[b * NNODE + j];

  const float sm = s[b * NNODE + i0 + m];   // already *log2e
  const int* adjRow = adj + ((size_t)(b * NNODE + i0 + m)) * NNODE;
  const _Float16* whB = whT + (size_t)b * FEAT * NNODE;

  v8f acc[4] = {v8f{}, v8f{}, v8f{}, v8f{}};
  float mrow = -__builtin_inff();
  float lsum = 0.f;

#if HAVE_TDM
  if (wave == 0) {
    tdm_load_wh(whB, (unsigned)(size_t)&ldsWh[0][0][0]);
    __builtin_amdgcn_s_wait_tensorcnt(0);
  }
  __syncthreads();  // chunk0 + ldsT visible
#endif

  for (int c = 0; c < NNODE / 32; ++c) {
    const int j0 = c * 32;
#if HAVE_TDM
    const int buf = c & 1;
    if (wave == 0 && c < NNODE / 32 - 1)
      tdm_load_wh(whB + (j0 + 32), (unsigned)(size_t)&ldsWh[buf ^ 1][0][0]);
#else
    const int buf = 0;
    __syncthreads();  // previous compute done (and ldsT ready at c==0)
    {
      const int n = tid >> 1, hf = tid & 1;
      const u4* sp = (const u4*)(whB + (size_t)n * NNODE + j0 + hf * 16);
      u4* dp = (u4*)&ldsWh[0][n][hf * 16];
      dp[0] = sp[0];
      dp[1] = sp[1];
    }
    __syncthreads();
#endif

    // --- adj (streaming, non-temporal) + e = mask(leakyrelu(s_i + t_j)) ---
    const i4* ap0 = (const i4*)(adjRow + j0 + off);
    const i4* ap1 = (const i4*)(adjRow + j0 + 16 + off);
    i4 a0 = __builtin_nontemporal_load(ap0);
    i4 a1 = __builtin_nontemporal_load(ap0 + 1);
    i4 a2 = __builtin_nontemporal_load(ap1);
    i4 a3 = __builtin_nontemporal_load(ap1 + 1);
    const f4* tp0 = (const f4*)&ldsT[j0 + off];
    const f4* tp1 = (const f4*)&ldsT[j0 + 16 + off];
    f4 t0 = tp0[0], t1 = tp0[1], t2 = tp1[0], t3 = tp1[1];

    float e[16];   // exp2-domain logits
#pragma unroll
    for (int q = 0; q < 4; ++q) {
      float x;
      x = sm + t0[q]; x = x > 0.f ? x : LRELU_A * x; e[q]      = a0[q] > 0 ? x : NEG_BIG;
      x = sm + t1[q]; x = x > 0.f ? x : LRELU_A * x; e[4 + q]  = a1[q] > 0 ? x : NEG_BIG;
      x = sm + t2[q]; x = x > 0.f ? x : LRELU_A * x; e[8 + q]  = a2[q] > 0 ? x : NEG_BIG;
      x = sm + t3[q]; x = x > 0.f ? x : LRELU_A * x; e[12 + q] = a3[q] > 0 ? x : NEG_BIG;
    }

    // --- online softmax (row stats shared by lane pair m, m+16) ---
    float cmax = e[0];
#pragma unroll
    for (int i = 1; i < 16; ++i) cmax = fmaxf(cmax, e[i]);
    cmax = fmaxf(cmax, xor16(cmax));
    const float nm = fmaxf(mrow, cmax);
    const float sc = fast_exp2(mrow - nm);
    mrow = nm;

    v16h af;
    float psum = 0.f;
#pragma unroll
    for (int i = 0; i < 16; ++i) {
      float p = fast_exp2(e[i] - nm);
      psum += p;
      af[i] = (_Float16)p;   // matches 16-bit A-fragment K ordering
    }
    lsum = lsum * sc + psum;

    // rescale accumulators only when the running max moved (wave-uniform test)
#if defined(__gfx1250__) && __has_builtin(__builtin_amdgcn_ballot_w32)
    if (__builtin_amdgcn_ballot_w32(sc < 1.0f))
#endif
    {
      const int base = hi << 3;   // C/D VGPR r holds output row r + 8*hi
#pragma unroll
      for (int r = 0; r < 8; ++r) {
        const float rs = lane_bcast(sc, r + base);
        acc[0][r] *= rs; acc[1][r] *= rs; acc[2][r] *= rs; acc[3][r] *= rs;
      }
    }

    // --- 4 WMMAs: p(16x32 f16) x WhT-tile(32x16 f16) -> 16x16 f32 ---
#pragma unroll
    for (int nt = 0; nt < 4; ++nt) {
      const h8* blo = (const h8*)&ldsWh[buf][nt * 16 + m][off];
      const h8* bhi = (const h8*)&ldsWh[buf][nt * 16 + m][16 + off];
      h8 lo = *blo, hh = *bhi;
      v16h bf = __builtin_shufflevector(lo, hh, 0, 1, 2, 3, 4, 5, 6, 7,
                                        8, 9, 10, 11, 12, 13, 14, 15);
      acc[nt] = __builtin_amdgcn_wmma_f32_16x16x32_f16(
          false, af, false, bf, (short)0, acc[nt], false, false);
    }

#if HAVE_TDM
    if (wave == 0) __builtin_amdgcn_s_wait_tensorcnt(0);
    __syncthreads();  // next buffer ready; this buffer free to overwrite
#endif
  }

  // --- epilogue: normalize, ELU, store ---
  const float ltot = lsum + xor16(lsum);  // full row sum (both lane halves)
  float* orow = out + ((size_t)(b * NNODE + i0)) * FEAT;
  const int base = hi << 3;
#pragma unroll
  for (int r = 0; r < 8; ++r) {
    const float inv = fast_rcp(lane_bcast(ltot, r + base));
    const int R = r + base;
#pragma unroll
    for (int nt = 0; nt < 4; ++nt) {
      float v = acc[nt][r] * inv;
      v = v > 0.f ? v : expm1f(v);
      orow[(size_t)R * FEAT + nt * 16 + m] = v;
    }
  }
}

extern "C" void kernel_launch(void* const* d_in, const int* in_sizes, int n_in,
                              void* d_out, int out_size, void* d_ws, size_t ws_size,
                              hipStream_t stream) {
  const float* h   = (const float*)d_in[0];
  const int*   adj = (const int*)d_in[1];
  const float* W   = (const float*)d_in[2];
  const float* a   = (const float*)d_in[3];
  float* out = (float*)d_out;

  char* ws = (char*)d_ws;
  _Float16* whT = (_Float16*)ws;                              // 2 MB
  float* s = (float*)(ws + 2u * 1024u * 1024u);               // 64 KB
  float* t = (float*)(ws + 2u * 1024u * 1024u + 64u * 1024u); // 64 KB

  gat_prep<<<dim3(BATCH * NNODE), dim3(FEAT), 0, stream>>>(h, W, a, whT, s, t);
  gat_main<<<dim3(BATCH * (NNODE / 64)), dim3(128), 0, stream>>>(adj, whT, s, t, out);
}
